// CombinedLossExp71_18064632446983
// MI455X (gfx1250) — matrix operand order repaired and checked by reading, not compiled, not run on hardware
//
#include <hip/hip_runtime.h>
#include <math.h>

// ---------------- problem constants ----------------
#define BB     8
#define TT     1500
#define DD     512
#define KK     4096
#define NTOK   (BB * TT)      // 12000
#define STRIDE 320
#define MARGIN 0.2f

typedef __attribute__((ext_vector_type(16))) __bf16 v16bf;
typedef __attribute__((ext_vector_type(8)))  __bf16 v8bf;
typedef __attribute__((ext_vector_type(8)))  float  v8f;

// =====================================================================
// Kernel 1: per-token prep. 8 waves/block, 1 wave per token.
//  - convert student/teacher fp32 -> bf16 (for WMMA)
//  - ||s||^2, ||t||^2 per token
//  - masked feature-loss and triplet-loss per-token values
// =====================================================================
__global__ __launch_bounds__(256)
void prep_tokens_kernel(const float* __restrict__ S, const float* __restrict__ T,
                        const int* __restrict__ lengths,
                        unsigned short* __restrict__ sbf_u, unsigned short* __restrict__ tbf_u,
                        float* __restrict__ x2s, float* __restrict__ x2t,
                        float* __restrict__ feat_tok, float* __restrict__ trip_tok)
{
    const int wave = threadIdx.x >> 5;
    const int lane = threadIdx.x & 31;
    const int n = blockIdx.x * 8 + wave;          // token id
    const int b = n / TT;
    const int t = n - b * TT;
    const int nb = (b + BB - 1) % BB;             // roll(shift=1): neg_teacher[b] = teacher[b-1]

    const float* sp = S + (size_t)n * DD;
    const float* tp = T + (size_t)n * DD;
    const float* np = T + ((size_t)nb * TT + t) * DD;
    __bf16* sbf = (__bf16*)sbf_u;
    __bf16* tbf = (__bf16*)tbf_u;

    float sd2 = 0.f, snd = 0.f, ss2 = 0.f, st2 = 0.f;
    #pragma unroll 4
    for (int d = lane; d < DD; d += 32) {
        float sv = sp[d], tv = tp[d], nv = np[d];
        float df = sv - tv; sd2 += df * df;
        float nd = sv - nv; snd += nd * nd;
        ss2 += sv * sv;
        st2 += tv * tv;
        sbf[(size_t)n * DD + d] = (__bf16)sv;
        tbf[(size_t)n * DD + d] = (__bf16)tv;
    }
    #pragma unroll
    for (int off = 16; off > 0; off >>= 1) {
        sd2 += __shfl_xor(sd2, off, 32);
        snd += __shfl_xor(snd, off, 32);
        ss2 += __shfl_xor(ss2, off, 32);
        st2 += __shfl_xor(st2, off, 32);
    }
    if (lane == 0) {
        x2s[n] = ss2;
        x2t[n] = st2;
        int valid = lengths[b] / STRIDE; if (valid > TT) valid = TT;
        float mk = (t < valid) ? 1.0f : 0.0f;
        feat_tok[n] = (sd2 * (1.0f / (float)DD)) * mk;
        trip_tok[n] = fmaxf(sqrtf(sd2) - sqrtf(snd) + MARGIN, 0.0f) * mk;
    }
}

// =====================================================================
// Kernel 2: codebook prep. 1 wave per code row.
// =====================================================================
__global__ __launch_bounds__(256)
void prep_codebook_kernel(const float* __restrict__ CB,
                          unsigned short* __restrict__ cbf_u, float* __restrict__ c2)
{
    const int wave = threadIdx.x >> 5;
    const int lane = threadIdx.x & 31;
    const int k = blockIdx.x * 8 + wave;
    const float* cp = CB + (size_t)k * DD;
    __bf16* cbf = (__bf16*)cbf_u;
    float s = 0.f;
    #pragma unroll 4
    for (int d = lane; d < DD; d += 32) {
        float v = cp[d];
        s += v * v;
        cbf[(size_t)k * DD + d] = (__bf16)v;
    }
    #pragma unroll
    for (int off = 16; off > 0; off >>= 1) s += __shfl_xor(s, off, 32);
    if (lane == 0) c2[k] = s;
}

// =====================================================================
// Kernel 3: fused GEMM (WMMA bf16) + streaming log-softmax over K codes.
// 16 tokens per block, 4 waves, each wave owns 1024 codes (64 N-tiles).
// Outputs per token: masked KL and min squared distance.
// =====================================================================
__global__ __launch_bounds__(128)
void soft_vq_kernel(const unsigned short* __restrict__ sbf_u,
                    const unsigned short* __restrict__ tbf_u,
                    const unsigned short* __restrict__ cbf_u,
                    const float* __restrict__ x2s, const float* __restrict__ x2t,
                    const float* __restrict__ c2p, const int* __restrict__ lengths,
                    float* __restrict__ kl_tok, float* __restrict__ mind_tok)
{
    __shared__ __bf16 lds_s[16][520];     // pad 8 bf16 -> conflict-free row access
    __shared__ __bf16 lds_t[16][520];
    __shared__ float  lx2s[16], lx2t[16];
    __shared__ float  lstats[4][16][5];   // per wave, per token-row: m_s, se_s, m_t, se_t, w

    const int tid  = threadIdx.x;
    const int wave = tid >> 5;
    const int lane = tid & 31;
    const int hf   = lane >> 4;           // lane half (0/1)
    const int l16  = lane & 15;
    const int tok0 = blockIdx.x * 16;

    const __bf16* sbf = (const __bf16*)sbf_u;
    const __bf16* tbf = (const __bf16*)tbf_u;
    const __bf16* cbf = (const __bf16*)cbf_u;

    // ---- stage 16x512 student/teacher bf16 tiles into LDS ----
    for (int idx = tid; idx < 1024; idx += 128) {     // 1024 uint4 per matrix
        int row = idx >> 6, col = idx & 63;
        *(uint4*)&lds_s[row][col * 8] = *(const uint4*)&sbf[((size_t)(tok0 + row)) * DD + col * 8];
        *(uint4*)&lds_t[row][col * 8] = *(const uint4*)&tbf[((size_t)(tok0 + row)) * DD + col * 8];
    }
    if (tid < 16)       lx2s[tid]      = x2s[tok0 + tid];
    else if (tid < 32)  lx2t[tid - 16] = x2t[tok0 + tid - 16];
    __syncthreads();

    // per-lane ||x||^2 for the rows this lane's accumulator slots cover
    float xs[8], xt[8];
    #pragma unroll
    for (int r = 0; r < 8; ++r) { xs[r] = lx2s[r + 8 * hf]; xt[r] = lx2t[r + 8 * hf]; }

    // streaming softmax state (per accumulator slot)
    float m_s[8], se_s[8], m_t[8], se_t[8], ww[8];
    #pragma unroll
    for (int r = 0; r < 8; ++r) { m_s[r] = -3.0e38f; se_s[r] = 0.f; m_t[r] = -3.0e38f; se_t[r] = 0.f; ww[r] = 0.f; }

    const __bf16* as0 = &lds_s[l16][hf * 8];   // A frag: VGPR0-3 -> K=e0..e0+7, VGPR4-7 -> K=e0+16..e0+23
    const __bf16* at0 = &lds_t[l16][hf * 8];

    #pragma unroll 1
    for (int tile = 0; tile < 64; ++tile) {
        const int code0 = wave * 1024 + tile * 16;
        const float c2v = c2p[code0 + l16];
        // B frag: lane = N column = codebook row; K = e0..e0+7 and e0+16..e0+23 (e0 = hf*8), matching A
        const __bf16* cbrow = cbf + (size_t)(code0 + l16) * DD + hf * 8;

        v8f acc_s = {0.f,0.f,0.f,0.f,0.f,0.f,0.f,0.f};
        v8f acc_t = {0.f,0.f,0.f,0.f,0.f,0.f,0.f,0.f};
        #pragma unroll
        for (int ch = 0; ch < 16; ++ch) {
            const int d0 = ch * 32;
            v8bf blo = *(const v8bf*)(cbrow + d0);
            v8bf bhi = *(const v8bf*)(cbrow + d0 + 16);
            v16bf bfrag = __builtin_shufflevector(blo, bhi, 0,1,2,3,4,5,6,7,8,9,10,11,12,13,14,15);
            v8bf slo = *(const v8bf*)(as0 + d0);
            v8bf shi = *(const v8bf*)(as0 + d0 + 16);
            v8bf tlo = *(const v8bf*)(at0 + d0);
            v8bf thi = *(const v8bf*)(at0 + d0 + 16);
            v16bf afs = __builtin_shufflevector(slo, shi, 0,1,2,3,4,5,6,7,8,9,10,11,12,13,14,15);
            v16bf aft = __builtin_shufflevector(tlo, thi, 0,1,2,3,4,5,6,7,8,9,10,11,12,13,14,15);
            acc_s = __builtin_amdgcn_wmma_f32_16x16x32_bf16(false, afs, false, bfrag, (short)0, acc_s, false, false);
            acc_t = __builtin_amdgcn_wmma_f32_16x16x32_bf16(false, aft, false, bfrag, (short)0, acc_t, false, false);
        }
        // epilogue: logits = -max(x2 + c2 - 2*xc, 0); online max/sumexp/weighted-sum
        #pragma unroll
        for (int r = 0; r < 8; ++r) {
            float ls = -fmaxf(xs[r] + c2v - 2.0f * acc_s[r], 0.0f);
            float lt = -fmaxf(xt[r] + c2v - 2.0f * acc_t[r], 0.0f);
            float nms = fmaxf(m_s[r], ls);
            se_s[r] = se_s[r] * __expf(m_s[r] - nms) + __expf(ls - nms);
            m_s[r] = nms;
            float nmt = fmaxf(m_t[r], lt);
            float ea = __expf(m_t[r] - nmt), eb = __expf(lt - nmt);
            se_t[r] = se_t[r] * ea + eb;
            ww[r]   = ww[r]   * ea + eb * (lt - ls);
            m_t[r] = nmt;
        }
    }

    // ---- merge across the 16 lanes of each half (each holds a disjoint code subset) ----
    #pragma unroll
    for (int r = 0; r < 8; ++r) {
        #pragma unroll
        for (int off = 8; off > 0; off >>= 1) {
            float om  = __shfl_xor(m_s[r], off, 16);
            float ose = __shfl_xor(se_s[r], off, 16);
            float nm  = fmaxf(m_s[r], om);
            se_s[r] = se_s[r] * __expf(m_s[r] - nm) + ose * __expf(om - nm);
            m_s[r] = nm;
            float omt  = __shfl_xor(m_t[r], off, 16);
            float oset = __shfl_xor(se_t[r], off, 16);
            float owv  = __shfl_xor(ww[r],  off, 16);
            float nmt  = fmaxf(m_t[r], omt);
            float ea = __expf(m_t[r] - nmt), eb = __expf(omt - nmt);
            se_t[r] = se_t[r] * ea + oset * eb;
            ww[r]   = ww[r]   * ea + owv  * eb;
            m_t[r] = nmt;
        }
    }
    if (l16 == 0) {
        #pragma unroll
        for (int r = 0; r < 8; ++r) {
            int row = r + 8 * hf;
            lstats[wave][row][0] = m_s[r];
            lstats[wave][row][1] = se_s[r];
            lstats[wave][row][2] = m_t[r];
            lstats[wave][row][3] = se_t[r];
            lstats[wave][row][4] = ww[r];
        }
    }
    __syncthreads();

    // ---- cross-wave merge (fixed order -> deterministic), finalize per token ----
    if (tid < 16) {
        float Ms = -3.0e38f, Ss = 0.f, Mt = -3.0e38f, St = 0.f, W = 0.f;
        for (int w = 0; w < 4; ++w) {
            float om = lstats[w][tid][0], ose = lstats[w][tid][1];
            float nm = fmaxf(Ms, om);
            Ss = Ss * __expf(Ms - nm) + ose * __expf(om - nm);
            Ms = nm;
            float omt = lstats[w][tid][2], oset = lstats[w][tid][3], owv = lstats[w][tid][4];
            float nmt = fmaxf(Mt, omt);
            float ea = __expf(Mt - nmt), eb = __expf(omt - nmt);
            St = St * ea + oset * eb;
            W  = W  * ea + owv  * eb;
            Mt = nmt;
        }
        // KL = E_t[t_logits - s_logits] + lse_s - lse_t
        float kl = W / St + (Ms + __logf(Ss)) - (Mt + __logf(St));
        int n = tok0 + tid;
        int b = n / TT, t = n - b * TT;
        int valid = lengths[b] / STRIDE; if (valid > TT) valid = TT;
        float mk = (t < valid) ? 1.0f : 0.0f;
        kl_tok[n]   = kl * mk;
        mind_tok[n] = -Ms;        // min squared distance (= commit/distort contribution)
    }
}

// =====================================================================
// Kernel 4: deterministic final reduction -> scalar loss
// =====================================================================
__global__ __launch_bounds__(256)
void finalize_kernel(const float* __restrict__ feat_tok, const float* __restrict__ trip_tok,
                     const float* __restrict__ kl_tok, const float* __restrict__ mind_tok,
                     const int* __restrict__ lengths, float* __restrict__ out)
{
    __shared__ float r0[256], r1[256], r2[256], r3[256];
    float s0 = 0.f, s1 = 0.f, s2 = 0.f, s3 = 0.f;
    for (int i = threadIdx.x; i < NTOK; i += 256) {
        s0 += feat_tok[i]; s1 += trip_tok[i]; s2 += kl_tok[i]; s3 += mind_tok[i];
    }
    r0[threadIdx.x] = s0; r1[threadIdx.x] = s1; r2[threadIdx.x] = s2; r3[threadIdx.x] = s3;
    __syncthreads();
    for (int off = 128; off > 0; off >>= 1) {
        if (threadIdx.x < off) {
            r0[threadIdx.x] += r0[threadIdx.x + off];
            r1[threadIdx.x] += r1[threadIdx.x + off];
            r2[threadIdx.x] += r2[threadIdx.x + off];
            r3[threadIdx.x] += r3[threadIdx.x + off];
        }
        __syncthreads();
    }
    if (threadIdx.x == 0) {
        float msum = 0.f;
        for (int b = 0; b < BB; ++b) {
            int v = lengths[b] / STRIDE; if (v > TT) v = TT;
            msum += (float)v;
        }
        // FEATURE_W=TRIPLET_W=SOFT_W=1; COMMIT_W=DISTORT_W=0.1, commit==distort numerically
        float total = (r0[0] + r1[0] + r2[0]) / msum
                    + 0.2f * r3[0] / ((float)NTOK * (float)DD);
        out[0] = total;
    }
}

// =====================================================================
// launcher
// =====================================================================
extern "C" void kernel_launch(void* const* d_in, const int* in_sizes, int n_in,
                              void* d_out, int out_size, void* d_ws, size_t ws_size,
                              hipStream_t stream)
{
    (void)in_sizes; (void)n_in; (void)out_size; (void)ws_size;
    const float* S  = (const float*)d_in[0];   // student_features (B,T,D) f32
    const float* T  = (const float*)d_in[1];   // teacher_features (B,T,D) f32
    /* d_in[2] = teacher_codes (unused by reference) */
    const float* CB = (const float*)d_in[3];   // codebook (K,D) f32
    const int* lengths = (const int*)d_in[4];  // (B,) i32
    float* out = (float*)d_out;

    char* ws = (char*)d_ws;
    size_t off = 0;
    auto take = [&](size_t bytes) -> char* {
        char* p = ws + off;
        off += (bytes + 255) & ~(size_t)255;
        return p;
    };
    unsigned short* sbf  = (unsigned short*)take((size_t)NTOK * DD * 2);   // 12.3 MB
    unsigned short* tbf  = (unsigned short*)take((size_t)NTOK * DD * 2);   // 12.3 MB
    unsigned short* cbf  = (unsigned short*)take((size_t)KK * DD * 2);     //  4.2 MB
    float* x2s      = (float*)take((size_t)NTOK * 4);
    float* x2t      = (float*)take((size_t)NTOK * 4);
    float* c2       = (float*)take((size_t)KK * 4);
    float* feat_tok = (float*)take((size_t)NTOK * 4);
    float* trip_tok = (float*)take((size_t)NTOK * 4);
    float* kl_tok   = (float*)take((size_t)NTOK * 4);
    float* mind_tok = (float*)take((size_t)NTOK * 4);

    prep_tokens_kernel<<<NTOK / 8, 256, 0, stream>>>(S, T, lengths, sbf, tbf,
                                                     x2s, x2t, feat_tok, trip_tok);
    prep_codebook_kernel<<<KK / 8, 256, 0, stream>>>(CB, cbf, c2);
    soft_vq_kernel<<<NTOK / 16, 128, 0, stream>>>(sbf, tbf, cbf, x2s, x2t, c2,
                                                  lengths, kl_tok, mind_tok);
    finalize_kernel<<<1, 256, 0, stream>>>(feat_tok, trip_tok, kl_tok, mind_tok, lengths, out);
}
